// Net_42331197670221
// MI455X (gfx1250) — compile-verified, matrix operand order checked by reading, not run on hardware
//
#include <hip/hip_runtime.h>
#include <hip/hip_bf16.h>

// ---------------- problem constants (match reference) ----------------
#define N_NODES 50000
#define E_EDGES 800000
#define D_IN    128
#define D_HID   256
#define G_NUM   512
#define LD      256            // row stride (elements) for node feature buffers
#define NL      ((long)N_NODES * LD)
#define TILES_M (N_NODES/16)   // 3125, exact
#define TILES_N (D_HID/16)     // 16
#define NGROUPS (TILES_M * 4)  // each wave does a 16x64 strip (4 N-tiles)
#define W_TOT   (256*128 + 4*256*256)  // 294912 elems per weight family

typedef __attribute__((ext_vector_type(16))) __bf16 v16bf;
typedef __attribute__((ext_vector_type(8)))  float  v8f;

// fp32 -> bf16 round-to-nearest-even
__device__ __forceinline__ __bf16 f2bf(float f) {
    union { float f; unsigned u; } v; v.f = f;
    unsigned r = (v.u + 0x7fffu + ((v.u >> 16) & 1u)) >> 16;
    union { unsigned short s; __bf16 b; } o; o.s = (unsigned short)r;
    return o.b;
}
__device__ __forceinline__ float bf2f_bits(unsigned short s) {
    union { unsigned u; float f; } o; o.u = ((unsigned)s) << 16;
    return o.f;
}

// Load one 16-element bf16 fragment: two contiguous 8-elem (16B) runs.
// (CDNA5 16-bit A/B layout: lane holds K=[kh*8..+7] and [16+kh*8..+7].)
__device__ __forceinline__ v16bf load_fragb(const __bf16* __restrict__ p0,
                                            const __bf16* __restrict__ p1) {
    union { uint4 q[2]; v16bf v; } u;
    u.q[0] = *(const uint4*)p0;
    u.q[1] = *(const uint4*)p1;
    return u.v;
}

// ---------------- kernels ----------------

__global__ void zero_f32(float4* __restrict__ p, long n4) {
    long i = (long)blockIdx.x * blockDim.x + threadIdx.x;
    if (i < n4) p[i] = make_float4(0.f, 0.f, 0.f, 0.f);
}

// generic fp32 -> bf16 conversion (weights; once per launch, tiny)
__global__ void cvt_f32_bf16(__bf16* __restrict__ dst, const float* __restrict__ src,
                             long n) {
    long i = (long)blockIdx.x * blockDim.x + threadIdx.x;
    if (i < n) dst[i] = f2bf(src[i]);
}

// fp32 buffer -> bf16 buffer, 8 elements/thread, 16B stores
__global__ void cvt_f32_bf16_x8(__bf16* __restrict__ dst, const float* __restrict__ src,
                                long n8) {
    long i = (long)blockIdx.x * blockDim.x + threadIdx.x;
    if (i >= n8) return;
    const float* s = src + i * 8;
    float4 a = *(const float4*)(s);
    float4 b = *(const float4*)(s + 4);
    union { unsigned short h[8]; uint4 q; } u;
    union { __bf16 b; unsigned short s; } c;
    c.b = f2bf(a.x); u.h[0] = c.s;  c.b = f2bf(a.y); u.h[1] = c.s;
    c.b = f2bf(a.z); u.h[2] = c.s;  c.b = f2bf(a.w); u.h[3] = c.s;
    c.b = f2bf(b.x); u.h[4] = c.s;  c.b = f2bf(b.y); u.h[5] = c.s;
    c.b = f2bf(b.z); u.h[6] = c.s;  c.b = f2bf(b.w); u.h[7] = c.s;
    *(uint4*)(dst + i * 8) = u.q;
}

// h[n, f] = sum_i atom_embs[i][x[n,i], f]  -> bf16
__global__ void atom_encoder(__bf16* __restrict__ h, const int* __restrict__ x,
                             const float* __restrict__ e0, const float* __restrict__ e1,
                             const float* __restrict__ e2, const float* __restrict__ e3,
                             const float* __restrict__ e4, const float* __restrict__ e5,
                             const float* __restrict__ e6, const float* __restrict__ e7,
                             const float* __restrict__ e8) {
    long idx = (long)blockIdx.x * blockDim.x + threadIdx.x;
    if (idx >= (long)N_NODES * D_IN) return;
    int n = (int)(idx >> 7);
    int f = (int)(idx & 127);
    const int* xr = x + (long)n * 9;
    float s = e0[(long)xr[0] * D_IN + f];
    s += e1[(long)xr[1] * D_IN + f];
    s += e2[(long)xr[2] * D_IN + f];
    s += e3[(long)xr[3] * D_IN + f];
    s += e4[(long)xr[4] * D_IN + f];
    s += e5[(long)xr[5] * D_IN + f];
    s += e6[(long)xr[6] * D_IN + f];
    s += e7[(long)xr[7] * D_IN + f];
    s += e8[(long)xr[8] * D_IN + f];
    h[(long)n * LD + f] = f2bf(s);
}

// m[dst[e], f..f+7] += h_bf16[src[e], f..f+7]  (fp32 accumulation via L2 atomics)
__global__ void scatter_add(float* __restrict__ m, const __bf16* __restrict__ h,
                            const int* __restrict__ ei, int din) {
    int chunks = din >> 3;
    long total = (long)E_EDGES * chunks;
    long i = (long)blockIdx.x * blockDim.x + threadIdx.x;
    if (i >= total) return;
    long e = i / chunks;
    int  f = (int)(i - e * chunks) << 3;
    int  s = ei[e];
    int  d = ei[E_EDGES + e];
    union { uint4 q; unsigned short hs[8]; } u;
    u.q = *(const uint4*)(h + (long)s * LD + f);
    float* dst = m + (long)d * LD + f;
    #pragma unroll
    for (int j = 0; j < 8; ++j) atomicAdd(dst + j, bf2f_bits(u.hs[j]));
}

// out_bf16[16x64 strip] = relu( m @ wr^T + h @ wo^T + br ), 4 WMMA tiles per wave
__global__ void gemm_dual_bias_relu(__bf16* __restrict__ out,
                                    const __bf16* __restrict__ mb,
                                    const __bf16* __restrict__ hb,
                                    const __bf16* __restrict__ wrb,
                                    const __bf16* __restrict__ wob,
                                    const float* __restrict__ br,
                                    int din) {
    int wgrp = blockIdx.x * (blockDim.x >> 5) + (threadIdx.x >> 5);
    if (wgrp >= NGROUPS) return;          // wave-uniform: EXEC stays all-1
    int tm  = wgrp >> 2;                  // M tile (16 rows)
    int tng = (wgrp & 3) * 4;             // first of 4 consecutive N tiles
    int lane = threadIdx.x & 31;
    int r16  = lane & 15;
    int kh   = lane >> 4;

    const __bf16* arow = mb + (long)(tm * 16 + r16) * LD + kh * 8;
    const __bf16* hrow = hb + (long)(tm * 16 + r16) * LD + kh * 8;
    const __bf16* wr0 = wrb + (long)((tng + 0) * 16 + r16) * din + kh * 8;
    const __bf16* wr1 = wrb + (long)((tng + 1) * 16 + r16) * din + kh * 8;
    const __bf16* wr2 = wrb + (long)((tng + 2) * 16 + r16) * din + kh * 8;
    const __bf16* wr3 = wrb + (long)((tng + 3) * 16 + r16) * din + kh * 8;
    const __bf16* wo0 = wob + (long)((tng + 0) * 16 + r16) * din + kh * 8;
    const __bf16* wo1 = wob + (long)((tng + 1) * 16 + r16) * din + kh * 8;
    const __bf16* wo2 = wob + (long)((tng + 2) * 16 + r16) * din + kh * 8;
    const __bf16* wo3 = wob + (long)((tng + 3) * 16 + r16) * din + kh * 8;

    v8f acc[4] = {{}, {}, {}, {}};

    #pragma unroll 2
    for (int k0 = 0; k0 < din; k0 += 32) {
        v16bf a = load_fragb(arow + k0, arow + k0 + 16);
        acc[0] = __builtin_amdgcn_wmma_f32_16x16x32_bf16(false, a, false,
                     load_fragb(wr0 + k0, wr0 + k0 + 16), (short)0, acc[0], false, false);
        acc[1] = __builtin_amdgcn_wmma_f32_16x16x32_bf16(false, a, false,
                     load_fragb(wr1 + k0, wr1 + k0 + 16), (short)0, acc[1], false, false);
        acc[2] = __builtin_amdgcn_wmma_f32_16x16x32_bf16(false, a, false,
                     load_fragb(wr2 + k0, wr2 + k0 + 16), (short)0, acc[2], false, false);
        acc[3] = __builtin_amdgcn_wmma_f32_16x16x32_bf16(false, a, false,
                     load_fragb(wr3 + k0, wr3 + k0 + 16), (short)0, acc[3], false, false);
    }
    #pragma unroll 2
    for (int k0 = 0; k0 < din; k0 += 32) {
        v16bf a = load_fragb(hrow + k0, hrow + k0 + 16);
        acc[0] = __builtin_amdgcn_wmma_f32_16x16x32_bf16(false, a, false,
                     load_fragb(wo0 + k0, wo0 + k0 + 16), (short)0, acc[0], false, false);
        acc[1] = __builtin_amdgcn_wmma_f32_16x16x32_bf16(false, a, false,
                     load_fragb(wo1 + k0, wo1 + k0 + 16), (short)0, acc[1], false, false);
        acc[2] = __builtin_amdgcn_wmma_f32_16x16x32_bf16(false, a, false,
                     load_fragb(wo2 + k0, wo2 + k0 + 16), (short)0, acc[2], false, false);
        acc[3] = __builtin_amdgcn_wmma_f32_16x16x32_bf16(false, a, false,
                     load_fragb(wo3 + k0, wo3 + k0 + 16), (short)0, acc[3], false, false);
    }

    // C/D layout: col = lane%16, row = vgpr + 8*(lane/16)
    long rbase = (long)(tm * 16 + kh * 8) * LD;
    #pragma unroll
    for (int j = 0; j < 4; ++j) {
        float bias = br[(tng + j) * 16 + r16];
        long obase = rbase + (tng + j) * 16 + r16;
        #pragma unroll
        for (int r = 0; r < 8; ++r) {
            float v = acc[j][r] + bias;
            out[obase + (long)r * LD] = f2bf(fmaxf(v, 0.0f));
        }
    }
}

// pool[batch[n], f..f+7] += h_bf16[n, f..f+7]; cnt[batch[n]] += 1
__global__ void pool_add(float* __restrict__ pool, float* __restrict__ cnt,
                         const __bf16* __restrict__ h, const int* __restrict__ batch) {
    long idx = (long)blockIdx.x * blockDim.x + threadIdx.x;
    if (idx >= (long)N_NODES * (D_HID / 8)) return;
    int n = (int)(idx >> 5);
    int f = (int)(idx & 31) << 3;
    int b = batch[n];
    union { uint4 q; unsigned short hs[8]; } u;
    u.q = *(const uint4*)(h + (long)n * LD + f);
    float* dst = pool + (long)b * D_HID + f;
    #pragma unroll
    for (int j = 0; j < 8; ++j) atomicAdd(dst + j, bf2f_bits(u.hs[j]));
    if (f == 0) atomicAdd(&cnt[b], 1.0f);
}

// out[g,:] = relu(mean_g @ w1^T + b1) @ w2^T + b2   (tiny: one thread per graph)
__global__ void mlp_head(float* __restrict__ out,
                         const float* __restrict__ pool, const float* __restrict__ cnt,
                         const float* __restrict__ w1, const float* __restrict__ b1,
                         const float* __restrict__ w2, const float* __restrict__ b2) {
    int g = blockIdx.x * blockDim.x + threadIdx.x;
    if (g >= G_NUM) return;
    float inv = 1.0f / fmaxf(cnt[g], 1.0f);
    const float* pg = pool + (long)g * D_HID;
    float hid[16];
    #pragma unroll
    for (int i = 0; i < 16; ++i) {
        float s = b1[i];
        const float* wrow = w1 + (long)i * D_HID;
        for (int k = 0; k < D_HID; ++k) s += pg[k] * inv * wrow[k];
        hid[i] = fmaxf(s, 0.0f);
    }
    #pragma unroll
    for (int o = 0; o < 2; ++o) {
        float s = b2[o];
        #pragma unroll
        for (int k = 0; k < 16; ++k) s += hid[k] * w2[o * 16 + k];
        out[(long)g * 2 + o] = s;
    }
}

// ---------------- host orchestration ----------------
extern "C" void kernel_launch(void* const* d_in, const int* in_sizes, int n_in,
                              void* d_out, int out_size, void* d_ws, size_t ws_size,
                              hipStream_t stream) {
    (void)in_sizes; (void)n_in; (void)out_size; (void)ws_size;

    // input layout per setup_inputs() flattening order
    const int*   x     = (const int*)d_in[0];
    const int*   ei    = (const int*)d_in[1];
    const int*   batch = (const int*)d_in[2];
    const float* embs[9];
    for (int i = 0; i < 9; ++i) embs[i] = (const float*)d_in[3 + i];
    const float* wr[5]; const float* br[5]; const float* wo[5];
    for (int i = 0; i < 5; ++i) {
        wr[i] = (const float*)d_in[12 + i];
        br[i] = (const float*)d_in[17 + i];
        wo[i] = (const float*)d_in[22 + i];
    }
    const float* w1 = (const float*)d_in[27];
    const float* b1 = (const float*)d_in[28];
    const float* w2 = (const float*)d_in[29];
    const float* b2 = (const float*)d_in[30];
    float* out = (float*)d_out;

    // workspace carve-out (all section sizes are multiples of 16 bytes)
    __bf16* hAb  = (__bf16*)d_ws;            // N*LD bf16
    __bf16* hBb  = hAb + NL;                 // N*LD bf16
    float*  mm   = (float*)(hBb + NL);       // N*LD f32
    __bf16* mb   = (__bf16*)(mm + NL);       // N*LD bf16
    __bf16* wrb  = mb + NL;                  // W_TOT bf16
    __bf16* wob  = wrb + W_TOT;              // W_TOT bf16
    float*  pool = (float*)(wob + W_TOT);    // G*256 f32
    float*  cnt  = pool + (long)G_NUM * D_HID;

    const int TB = 256;
    const int dins[5]  = { D_IN, D_HID, D_HID, D_HID, D_HID };
    long woff[5]; woff[0] = 0;
    for (int l = 1; l < 5; ++l) woff[l] = woff[l-1] + (long)dins[l-1] * D_HID;

    // 0) weights -> bf16 (once per launch; tiny)
    for (int l = 0; l < 5; ++l) {
        long n = (long)dins[l] * D_HID;
        int blocks = (int)((n + TB - 1) / TB);
        cvt_f32_bf16<<<blocks, TB, 0, stream>>>(wrb + woff[l], wr[l], n);
        cvt_f32_bf16<<<blocks, TB, 0, stream>>>(wob + woff[l], wo[l], n);
    }

    // 1) atom encoder -> bf16 h
    {
        long tot = (long)N_NODES * D_IN;
        atom_encoder<<<(int)((tot + TB - 1) / TB), TB, 0, stream>>>(
            hAb, x, embs[0], embs[1], embs[2], embs[3], embs[4],
            embs[5], embs[6], embs[7], embs[8]);
    }

    // 2) GraphConv stack
    for (int l = 0; l < 5; ++l) {
        int din = dins[l];
        {   // zero fp32 message accumulator
            long n4 = NL / 4;
            zero_f32<<<(int)((n4 + TB - 1) / TB), TB, 0, stream>>>((float4*)mm, n4);
        }
        {   // edge scatter-add (bf16 gather -> fp32 atomics)
            long tot = (long)E_EDGES * (din >> 3);
            scatter_add<<<(int)((tot + TB - 1) / TB), TB, 0, stream>>>(mm, hAb, ei, din);
        }
        {   // m -> bf16 (one pass)
            long n8 = NL / 8;
            cvt_f32_bf16_x8<<<(int)((n8 + TB - 1) / TB), TB, 0, stream>>>(mb, mm, n8);
        }
        {   // WMMA GEMM strip kernel
            int blocks = (NGROUPS + (TB / 32) - 1) / (TB / 32);   // 12500/8 -> 1563
            gemm_dual_bias_relu<<<blocks, TB, 0, stream>>>(
                hBb, mb, hAb, wrb + woff[l], wob + woff[l], br[l], din);
        }
        __bf16* t = hAb; hAb = hBb; hBb = t;
    }

    // 3) mean pool
    {
        long n4 = ((long)G_NUM * D_HID + G_NUM) / 4;   // pool + cnt contiguous
        zero_f32<<<(int)((n4 + TB - 1) / TB), TB, 0, stream>>>((float4*)pool, n4);
        long tot = (long)N_NODES * (D_HID / 8);
        pool_add<<<(int)((tot + TB - 1) / TB), TB, 0, stream>>>(pool, cnt, hAb, batch);
    }

    // 4) MLP head
    mlp_head<<<(G_NUM + TB - 1) / TB, TB, 0, stream>>>(out, pool, cnt, w1, b1, w2, b2);
}
